// PoseInitializer_34608846471342
// MI455X (gfx1250) — matrix operand order repaired and checked by reading, not compiled, not run on hardware
//
// MI455X / gfx1250 implementation of the GATv2 + triangle-voting pose initializer.
// Bandwidth-bound pipeline (round-0 roofline): bf16 WMMA 16x16x32 GEMMs with f32
// accumulate; L2-resident atomics for segment softmax/aggregation; triangle gather
// fused into the WMMA A-operand fetch; per-lane 3x3 Jacobi-Kabsch; LDS-tiled
// O(T^2) Gaussian voting. Epilogues templated (no runtime branches), fast v_rcp
// sigmoid, global_prefetch_b8 on the streamed A operand.
#include <hip/hip_runtime.h>
#include <hip/hip_bf16.h>

#define BB   4
#define NN   2048
#define TT   4096
#define BNN  (BB * NN)          // 8192 nodes
#define BTT  (BB * TT)          // 16384 triangles
#define EE   (BTT * 6)          // 98304 directed edges
#define SLOPE 0.2f
#define INV2SIG (-0.125f)       // -1/(2*sigma^2), sigma=2

typedef __bf16 bf16_t;
typedef bf16_t v16bf __attribute__((ext_vector_type(16)));
typedef float  v8f   __attribute__((ext_vector_type(8)));

// ---------- small device helpers ----------
__device__ __forceinline__ float fast_rcp(float x) {
#if __has_builtin(__builtin_amdgcn_rcpf)
    return __builtin_amdgcn_rcpf(x);        // v_rcp_f32
#else
    return 1.0f / x;
#endif
}
__device__ __forceinline__ float sigmoidf_(float x) { return fast_rcp(1.0f + __expf(-x)); }
__device__ __forceinline__ float siluf_(float x)    { return x * sigmoidf_(x); }

// order-preserving float<->uint map for atomicMax on floats
__device__ __forceinline__ unsigned f32_ord(float f) {
    unsigned u = __float_as_uint(f);
    return (u & 0x80000000u) ? ~u : (u | 0x80000000u);
}
__device__ __forceinline__ float ord_f32(unsigned u) {
    unsigned b = (u & 0x80000000u) ? (u & 0x7FFFFFFFu) : ~u;
    return __uint_as_float(b);
}
#define ORD_NEG_INF 0x007FFFFFu   // f32_ord(-inf)

__device__ __forceinline__ float wave_sum32(float v) {
    v += __shfl_xor(v, 1, 32);
    v += __shfl_xor(v, 2, 32);
    v += __shfl_xor(v, 4, 32);
    v += __shfl_xor(v, 8, 32);
    v += __shfl_xor(v, 16, 32);
    return v;
}

// ---------- WMMA fragment packing ----------
// ISA 16-bit A 16x32 layout: lane's 16 elements are K-offsets {0..7} and {16..23}
// relative to (k0 + 8*hi) within the row -> four b128 loads + cvt_pk_bf16.
__device__ __forceinline__ v16bf pack_a(const float* __restrict__ p) {
    float4 x0 = *(const float4*)(p);
    float4 x1 = *(const float4*)(p + 4);
    float4 y0 = *(const float4*)(p + 16);
    float4 y1 = *(const float4*)(p + 20);
    v16bf a;
    a[0] = (bf16_t)x0.x;  a[1] = (bf16_t)x0.y;  a[2] = (bf16_t)x0.z;  a[3] = (bf16_t)x0.w;
    a[4] = (bf16_t)x1.x;  a[5] = (bf16_t)x1.y;  a[6] = (bf16_t)x1.z;  a[7] = (bf16_t)x1.w;
    a[8] = (bf16_t)y0.x;  a[9] = (bf16_t)y0.y;  a[10] = (bf16_t)y0.z; a[11] = (bf16_t)y0.w;
    a[12] = (bf16_t)y1.x; a[13] = (bf16_t)y1.y; a[14] = (bf16_t)y1.z; a[15] = (bf16_t)y1.w;
    return a;
}
// ISA 16-bit B 32x16 layout: element e at row (k0 + 8*hi + e + 8*(e>=8)), fixed column.
__device__ __forceinline__ v16bf pack_b(const float* __restrict__ p, int stride) {
    v16bf b;
#pragma unroll
    for (int e = 0; e < 16; ++e) b[e] = (bf16_t)p[(size_t)(e + (e >> 3) * 8) * stride];
    return b;
}

// ---------- 3x3 Kabsch via Jacobi eigensolve of K^T K ----------
__device__ void kabsch3(const float Kin[9], float R[9]) {
    float K[3][3];
    for (int i = 0; i < 3; ++i)
        for (int j = 0; j < 3; ++j) K[i][j] = Kin[i * 3 + j];
    float S[3][3];
    for (int i = 0; i < 3; ++i)
        for (int j = 0; j < 3; ++j)
            S[i][j] = K[0][i] * K[0][j] + K[1][i] * K[1][j] + K[2][i] * K[2][j];
    float V[3][3] = {{1.f, 0.f, 0.f}, {0.f, 1.f, 0.f}, {0.f, 0.f, 1.f}};
    const int pp[3] = {0, 0, 1}, qq[3] = {1, 2, 2};
    for (int sweep = 0; sweep < 8; ++sweep) {
        for (int r = 0; r < 3; ++r) {
            int p = pp[r], q = qq[r];
            float apq = S[p][q];
            if (fabsf(apq) < 1e-14f) continue;
            float tau = (S[q][q] - S[p][p]) / (2.0f * apq);
            float t = (tau >= 0.f ? 1.f : -1.f) / (fabsf(tau) + sqrtf(1.f + tau * tau));
            float c = rsqrtf(1.f + t * t), s = t * c;
            for (int k = 0; k < 3; ++k) { float a = S[k][p], b = S[k][q]; S[k][p] = c * a - s * b; S[k][q] = s * a + c * b; }
            for (int k = 0; k < 3; ++k) { float a = S[p][k], b = S[q][k]; S[p][k] = c * a - s * b; S[q][k] = s * a + c * b; }
            for (int k = 0; k < 3; ++k) { float a = V[k][p], b = V[k][q]; V[k][p] = c * a - s * b; V[k][q] = s * a + c * b; }
        }
    }
    float lam[3] = {S[0][0], S[1][1], S[2][2]};
    for (int a = 0; a < 2; ++a)
        for (int b2 = a + 1; b2 < 3; ++b2)
            if (lam[b2] > lam[a]) {
                float tl = lam[a]; lam[a] = lam[b2]; lam[b2] = tl;
                for (int k = 0; k < 3; ++k) { float tv = V[k][a]; V[k][a] = V[k][b2]; V[k][b2] = tv; }
            }
    // U columns: u0 = normalize(K v0); u1 = orthonormalized K v1; u2 = u0 x u1 (det(U)=+1)
    float u0[3], u1[3], u2[3];
    for (int i = 0; i < 3; ++i) u0[i] = K[i][0] * V[0][0] + K[i][1] * V[1][0] + K[i][2] * V[2][0];
    float n0 = rsqrtf(u0[0] * u0[0] + u0[1] * u0[1] + u0[2] * u0[2] + 1e-24f);
    for (int i = 0; i < 3; ++i) u0[i] *= n0;
    for (int i = 0; i < 3; ++i) u1[i] = K[i][0] * V[0][1] + K[i][1] * V[1][1] + K[i][2] * V[2][1];
    float d01 = u1[0] * u0[0] + u1[1] * u0[1] + u1[2] * u0[2];
    for (int i = 0; i < 3; ++i) u1[i] -= d01 * u0[i];
    float n1 = rsqrtf(u1[0] * u1[0] + u1[1] * u1[1] + u1[2] * u1[2] + 1e-24f);
    for (int i = 0; i < 3; ++i) u1[i] *= n1;
    u2[0] = u0[1] * u1[2] - u0[2] * u1[1];
    u2[1] = u0[2] * u1[0] - u0[0] * u1[2];
    u2[2] = u0[0] * u1[1] - u0[1] * u1[0];
    float dV = V[0][0] * (V[1][1] * V[2][2] - V[1][2] * V[2][1])
             - V[0][1] * (V[1][0] * V[2][2] - V[1][2] * V[2][0])
             + V[0][2] * (V[1][0] * V[2][1] - V[1][1] * V[2][0]);
    // R = V * diag(1,1,det(V U^T)) * U^T ; det(U)=+1 so d = det(V)
    float U[3][3];
    for (int j = 0; j < 3; ++j) { U[j][0] = u0[j]; U[j][1] = u1[j]; U[j][2] = u2[j]; }
    for (int i = 0; i < 3; ++i)
        for (int j = 0; j < 3; ++j)
            R[i * 3 + j] = V[i][0] * U[j][0] + V[i][1] * U[j][1] + dV * V[i][2] * U[j][2];
}

// ---------- fills ----------
__global__ void fill_f32_k(float* p, float v, int n) { int i = blockIdx.x * blockDim.x + threadIdx.x; if (i < n) p[i] = v; }
__global__ void fill_u32_k(unsigned* p, unsigned v, int n) { int i = blockIdx.x * blockDim.x + threadIdx.x; if (i < n) p[i] = v; }

// ---------- xc assembly: descs copy + positional encoder ----------
__global__ void copy_desc_k(const float* __restrict__ descs, float* __restrict__ xc) {
    int i = blockIdx.x * blockDim.x + threadIdx.x;
    if (i < BNN * 256) xc[(size_t)(i >> 8) * 320 + (i & 255)] = descs[i];
}

__global__ void posenc_k(const float* __restrict__ kpts, const float* __restrict__ pts3d,
                         const float* __restrict__ l1w, const float* __restrict__ l1b,
                         const float* __restrict__ ln1g, const float* __restrict__ ln1b,
                         const float* __restrict__ l2w, const float* __restrict__ l2b,
                         const float* __restrict__ ln2g, const float* __restrict__ ln2b,
                         float* __restrict__ xc) {
    int n = blockIdx.x * blockDim.x + threadIdx.x;
    if (n >= BNN) return;
    float in0 = kpts[n * 2 + 0] * (1.0f / 1216.0f);
    float in1 = kpts[n * 2 + 1] * (1.0f / 352.0f);
    float dz = pts3d[n * 3 + 2];
    float in2 = fminf(fmaxf(dz, 0.1f), 100.0f);
    float h[64];
    // l1 (3->32) + layernorm
    float m = 0.f;
    for (int o = 0; o < 32; ++o) { float v = in0 * l1w[o] + in1 * l1w[32 + o] + in2 * l1w[64 + o] + l1b[o]; h[o] = v; m += v; }
    m *= (1.f / 32.f);
    float var = 0.f;
    for (int o = 0; o < 32; ++o) { float d = h[o] - m; var += d * d; }
    var *= (1.f / 32.f);
    float rs = rsqrtf(var + 1e-5f);
    for (int o = 0; o < 32; ++o) h[o] = (h[o] - m) * rs * ln1g[o] + ln1b[o];
    // silu -> l2 (32->64) + layernorm
    float g[64];
    for (int o = 0; o < 64; ++o) g[o] = l2b[o];
    for (int i = 0; i < 32; ++i) {
        float si = siluf_(h[i]);
        for (int o = 0; o < 64; ++o) g[o] += si * l2w[i * 64 + o];
    }
    m = 0.f;
    for (int o = 0; o < 64; ++o) m += g[o];
    m *= (1.f / 64.f);
    var = 0.f;
    for (int o = 0; o < 64; ++o) { float d = g[o] - m; var += d * d; }
    var *= (1.f / 64.f);
    rs = rsqrtf(var + 1e-5f);
    for (int o = 0; o < 64; ++o) xc[(size_t)n * 320 + 256 + o] = (g[o] - m) * rs * ln2g[o] + ln2b[o];
}

// ---------- edge build: directed edges, edge_attr, outputs ----------
__global__ void edge_build_k(const int* __restrict__ tri, const float* __restrict__ kpts,
                             float* __restrict__ out_edges, float* __restrict__ out_eattr,
                             int* __restrict__ srcdst) {
    int tt = blockIdx.x * blockDim.x + threadIdx.x;
    if (tt >= BTT) return;
    int b = tt / TT;
    int gi = b * NN + tri[tt * 3 + 0];
    int gj = b * NN + tri[tt * 3 + 1];
    int gk = b * NN + tri[tt * 3 + 2];
    int sL[6] = {gi, gj, gk, gj, gk, gi};
    int dL[6] = {gj, gk, gi, gi, gj, gk};
    for (int c = 0; c < 6; ++c) {
        int e = tt * 6 + c;
        int s = sL[c], d = dL[c];
        srcdst[e * 2 + 0] = s;
        srcdst[e * 2 + 1] = d;
        out_edges[e] = (float)s;            // edges[0, :]
        out_edges[EE + e] = (float)d;       // edges[1, :]
        float sx = kpts[s * 2] * (1.0f / 1216.0f), sy = kpts[s * 2 + 1] * (1.0f / 352.0f);
        float dx = kpts[d * 2] * (1.0f / 1216.0f), dy = kpts[d * 2 + 1] * (1.0f / 352.0f);
        float rx = dx - sx, ry = dy - sy;
        out_eattr[e * 3 + 0] = rx;
        out_eattr[e * 3 + 1] = ry;
        out_eattr[e * 3 + 2] = sqrtf(rx * rx + ry * ry);
    }
}

// ---------- dense bf16 WMMA GEMM: C[M,N] = act(A[M,K] @ W[K,N] [+ bias]) ----------
// ACT: 0 = none, 1 = silu. HAS_BIAS compile-time -> branch-free epilogue.
template <int ACT, bool HAS_BIAS>
__global__ void gemm_wmma_dense_k(const float* __restrict__ A, const float* __restrict__ W,
                                  const float* __restrict__ bias, float* __restrict__ C,
                                  int M, int N, int K) {
    int wave = (int)((blockIdx.x * blockDim.x + threadIdx.x) >> 5);
    int lane = threadIdx.x & 31;
    int ntile = N >> 4;
    int tiles = (M >> 4) * ntile;
    if (wave >= tiles) return;               // launches use exact multiples: no partial waves
    int tm = wave / ntile, tn = wave % ntile;
    int l15 = lane & 15;
    int hi = lane >> 4;                      // +8 to A-K, +8 to B-K
    int colB = tn * 16 + l15;
    const float* abase = A + (size_t)(tm * 16 + l15) * K + hi * 8;
    const float* wbase = W + (size_t)(hi * 8) * N + colB;
    v8f c = {};
    for (int k0 = 0; k0 < K; k0 += 32) {
        __builtin_prefetch(abase + k0 + 32, 0, 0);   // global_prefetch_b8: next A chunk
        v16bf a = pack_a(abase + k0);
        v16bf b = pack_b(wbase + (size_t)k0 * N, N);
        c = __builtin_amdgcn_wmma_f32_16x16x32_bf16(false, a, false, b, (short)0, c, false, false);
    }
    float bv = HAS_BIAS ? bias[colB] : 0.0f;
    float* crow = C + (size_t)(tm * 16 + hi * 8) * N + colB;
#pragma unroll
    for (int r = 0; r < 8; ++r) {
        float v = c[r] + bv;
        if (ACT == 1) v = siluf_(v);
        crow[(size_t)r * N] = v;
    }
}

// ---------- triangle-head GEMM with fused vertex gather ----------
// h1 = silu(gather(nf, tri) @ W + bias); A-row r is triangle r, K=768 = 3 x 256.
// K-chunks of 32 never straddle a vertex block (256 % 32 == 0) -> one scalar
// vertex select per chunk, then the same vectorized A-fragment loads.
__global__ void gemm_wmma_gather_k(const float* __restrict__ NF, const int* __restrict__ tri,
                                   const float* __restrict__ W, const float* __restrict__ bias,
                                   float* __restrict__ C) {
    const int N = 128, K = 768;
    int wave = (int)((blockIdx.x * blockDim.x + threadIdx.x) >> 5);
    int lane = threadIdx.x & 31;
    int ntile = N >> 4;                      // 8
    int tiles = (BTT >> 4) * ntile;          // 8192
    if (wave >= tiles) return;
    int tm = wave / ntile, tn = wave % ntile;
    int l15 = lane & 15;
    int hi = lane >> 4;
    int rowA = tm * 16 + l15;
    int colB = tn * 16 + l15;
    int t0 = tri[rowA * 3 + 0], t1 = tri[rowA * 3 + 1], t2 = tri[rowA * 3 + 2];
    int gb = (rowA / TT) * NN;
    const float* vb0 = NF + (size_t)(gb + t0) * 256 + hi * 8;
    const float* vb1 = NF + (size_t)(gb + t1) * 256 + hi * 8;
    const float* vb2 = NF + (size_t)(gb + t2) * 256 + hi * 8;
    const float* wbase = W + (size_t)(hi * 8) * N + colB;
    v8f c = {};
    for (int k0 = 0; k0 < K; k0 += 32) {
        int v = k0 >> 8;                     // scalar: uniform vertex per chunk
        const float* ab = (v == 0) ? vb0 : ((v == 1) ? vb1 : vb2);
        v16bf a = pack_a(ab + (k0 & 255));
        v16bf b = pack_b(wbase + (size_t)k0 * N, N);
        c = __builtin_amdgcn_wmma_f32_16x16x32_bf16(false, a, false, b, (short)0, c, false, false);
    }
    float bv = bias[colB];
    float* crow = C + (size_t)(tm * 16 + hi * 8) * N + colB;
#pragma unroll
    for (int r = 0; r < 8; ++r) {
        crow[(size_t)r * N] = siluf_(c[r] + bv);
    }
}

// ---------- GATv2 attention logits + segment max ----------
__global__ void edge_e_k(const float* __restrict__ xl, const float* __restrict__ xr,
                         const float* __restrict__ eattr, const float* __restrict__ we,
                         const float* __restrict__ att, const int* __restrict__ srcdst,
                         float* __restrict__ evals, unsigned* __restrict__ mmax) {
    int e = (int)((blockIdx.x * blockDim.x + threadIdx.x) >> 5);
    int lane = threadIdx.x & 31;
    if (e >= EE) return;
    int s = srcdst[e * 2], d = srcdst[e * 2 + 1];
    float ea0 = eattr[e * 3], ea1 = eattr[e * 3 + 1], ea2 = eattr[e * 3 + 2];
    int h = lane >> 3;                       // 8 lanes per head, 8 dims per lane
    float part = 0.f;
#pragma unroll
    for (int q = 0; q < 8; ++q) {
        int dim = lane * 8 + q;
        float ee = ea0 * we[dim] + ea1 * we[256 + dim] + ea2 * we[512 + dim];
        float z = xl[(size_t)s * 256 + dim] + xr[(size_t)d * 256 + dim] + ee;
        z = (z > 0.f) ? z : SLOPE * z;       // leaky_relu
        part += z * att[h * 64 + (dim & 63)];
    }
    part += __shfl_xor(part, 1, 32);
    part += __shfl_xor(part, 2, 32);
    part += __shfl_xor(part, 4, 32);
    if ((lane & 7) == 0) {
        evals[e * 4 + h] = part;
        atomicMax(&mmax[(size_t)d * 4 + h], f32_ord(part));
    }
}

// ---------- exp(e - max) and segment sum ----------
__global__ void edge_softmax_k(const float* __restrict__ evals, const unsigned* __restrict__ mmax,
                               const int* __restrict__ srcdst, float* __restrict__ exvals,
                               float* __restrict__ ssum) {
    int idx = blockIdx.x * blockDim.x + threadIdx.x;
    if (idx >= EE * 4) return;
    int e = idx >> 2, h = idx & 3;
    int d = srcdst[e * 2 + 1];
    float ex = __expf(evals[idx] - ord_f32(mmax[(size_t)d * 4 + h]));
    exvals[idx] = ex;
    atomicAdd(&ssum[(size_t)d * 4 + h], ex);
}

// ---------- weighted aggregation: out[dst] += alpha * xl[src] ----------
__global__ void edge_agg_k(const float* __restrict__ xl, const float* __restrict__ exvals,
                           const float* __restrict__ ssum, const int* __restrict__ srcdst,
                           float* __restrict__ agg) {
    int e = (int)((blockIdx.x * blockDim.x + threadIdx.x) >> 5);
    int lane = threadIdx.x & 31;
    if (e >= EE) return;
    int s = srcdst[e * 2], d = srcdst[e * 2 + 1];
    int h = lane >> 3;
    float alpha = exvals[e * 4 + h] * fast_rcp(ssum[(size_t)d * 4 + h] + 1e-16f);
#pragma unroll
    for (int q = 0; q < 8; ++q) {
        int dim = lane * 8 + q;
        atomicAdd(&agg[(size_t)d * 256 + dim], alpha * xl[(size_t)s * 256 + dim]);
    }
}

// ---------- node post: layernorm(agg + bias); silu(y + res_proj) ----------
__global__ void node_post_k(const float* __restrict__ agg, const float* __restrict__ gbias,
                            const float* __restrict__ ng, const float* __restrict__ nb,
                            const float* __restrict__ rp, float* __restrict__ tb) {
    int n = (int)((blockIdx.x * blockDim.x + threadIdx.x) >> 5);
    int lane = threadIdx.x & 31;
    if (n >= BNN) return;
    float v[8];
    float sum = 0.f, sq = 0.f;
#pragma unroll
    for (int q = 0; q < 8; ++q) {
        int d = q * 32 + lane;
        float x = agg[(size_t)n * 256 + d] + gbias[d];
        v[q] = x; sum += x; sq += x * x;
    }
    sum = wave_sum32(sum); sq = wave_sum32(sq);
    float mean = sum * (1.f / 256.f);
    float var = sq * (1.f / 256.f) - mean * mean;
    float rs = rsqrtf(var + 1e-5f);
#pragma unroll
    for (int q = 0; q < 8; ++q) {
        int d = q * 32 + lane;
        float y = (v[q] - mean) * rs * ng[d] + nb[d];
        tb[(size_t)n * 256 + d] = siluf_(y + rp[(size_t)n * 256 + d]);
    }
}

// ---------- triangle weight head output: sigmoid(h2 @ whw + whb) ----------
__global__ void dot_wh_k(const float* __restrict__ h2, const float* __restrict__ whw,
                         const float* __restrict__ whb, float* __restrict__ wtri) {
    int r = (int)((blockIdx.x * blockDim.x + threadIdx.x) >> 5);
    int lane = threadIdx.x & 31;
    if (r >= BTT) return;
    float part = 0.f;
#pragma unroll
    for (int q = 0; q < 4; ++q) {
        int d = q * 32 + lane;
        part += h2[(size_t)r * 128 + d] * whw[d];
    }
    part = wave_sum32(part);
    if (lane == 0) wtri[r] = sigmoidf_(part + whb[0]);
}

// ---------- per-triangle correlation K, Kabsch R, vanishing-x vote ----------
// note: reference adds N(0, 1e-5) jax-PRNG noise to K; stream not reproducible
// on-device and numerically negligible -> omitted.
__global__ void tri_K_k(const int* __restrict__ tri, const float* __restrict__ pts3d,
                        const float* __restrict__ kt1, const float* __restrict__ intr,
                        float* __restrict__ Ktri, float* __restrict__ xv) {
    int r = blockIdx.x * blockDim.x + threadIdx.x;
    if (r >= BTT) return;
    int b = r / TT;
    float fx = intr[b * 4], fy = intr[b * 4 + 1], cx = intr[b * 4 + 2], cy = intr[b * 4 + 3];
    float Km[9];
#pragma unroll
    for (int i = 0; i < 9; ++i) Km[i] = 0.f;
    for (int v = 0; v < 3; ++v) {
        int nd = b * NN + tri[r * 3 + v];
        float p0 = pts3d[nd * 3], p1 = pts3d[nd * 3 + 1], p2 = pts3d[nd * 3 + 2];
        float inv = fast_rcp(sqrtf(p0 * p0 + p1 * p1 + p2 * p2) + 1e-8f);
        float a0 = p0 * inv, a1 = p1 * inv, a2 = p2 * inv;
        float bx = (kt1[nd * 2] - cx) / (fx + 1e-8f);
        float by = (kt1[nd * 2 + 1] - cy) / (fy + 1e-8f);
        Km[0] += a0 * bx; Km[1] += a0 * by; Km[2] += a0;
        Km[3] += a1 * bx; Km[4] += a1 * by; Km[5] += a1;
        Km[6] += a2 * bx; Km[7] += a2 * by; Km[8] += a2;
    }
    float R[9];
    kabsch3(Km, R);
    xv[r] = fx * (R[2] * fast_rcp(R[8] + 1e-8f)) + cx;   // fx * R[0,2]/R[2,2] + cx
#pragma unroll
    for (int i = 0; i < 9; ++i) Ktri[r * 9 + i] = Km[i];
}

// ---------- Gaussian voting density: dens_i = sum_j exp(-(xi-xj)^2/8) w_j ----------
__global__ void dens_k(const float* __restrict__ xv, const float* __restrict__ wtri,
                       float* __restrict__ dens) {
    int b = blockIdx.y;
    int i = blockIdx.x * 256 + threadIdx.x;
    float xi = xv[b * TT + i];
    __shared__ float sx[256], sw[256];
    float acc = 0.f;
    for (int j0 = 0; j0 < TT; j0 += 256) {
        sx[threadIdx.x] = xv[b * TT + j0 + threadIdx.x];
        sw[threadIdx.x] = wtri[b * TT + j0 + threadIdx.x];
        __syncthreads();
        for (int jj = 0; jj < 256; ++jj) {
            float dx = xi - sx[jj];
            acc += __expf(dx * dx * INV2SIG) * sw[jj];
        }
        __syncthreads();
    }
    dens[b * TT + i] = acc;
}

// ---------- weighted vote mean xv_star per batch ----------
__global__ void xvstar_k(const float* __restrict__ xv, const float* __restrict__ wtri,
                         const float* __restrict__ dens, float* __restrict__ xvstar) {
    int b = blockIdx.x;
    __shared__ float sn[256], sd[256];
    float num = 0.f, den = 0.f;
    for (int i = threadIdx.x; i < TT; i += 256) {
        float vw = wtri[b * TT + i] * dens[b * TT + i];
        num += vw * xv[b * TT + i];
        den += vw;
    }
    sn[threadIdx.x] = num; sd[threadIdx.x] = den;
    __syncthreads();
    for (int s = 128; s > 0; s >>= 1) {
        if (threadIdx.x < s) { sn[threadIdx.x] += sn[threadIdx.x + s]; sd[threadIdx.x] += sd[threadIdx.x + s]; }
        __syncthreads();
    }
    if (threadIdx.x == 0) xvstar[b] = sn[0] / (sd[0] + 1e-8f);
}

// ---------- consensus weights, global K accumulation, conf scatter ----------
__global__ void cw_k(const int* __restrict__ tri, const float* __restrict__ xv,
                     const float* __restrict__ xvstar, const float* __restrict__ wtri,
                     const float* __restrict__ Ktri, float* __restrict__ out_cw,
                     float* __restrict__ Kg, float* __restrict__ confb) {
    int r = blockIdx.x * blockDim.x + threadIdx.x;
    if (r >= BTT) return;
    int b = r / TT;
    float diff = xv[r] - xvstar[b];
    float s = __expf(diff * diff * INV2SIG);
    float cwv = wtri[r] * s;
    out_cw[r] = cwv;
#pragma unroll
    for (int i = 0; i < 9; ++i) atomicAdd(&Kg[b * 9 + i], cwv * Ktri[r * 9 + i]);
    for (int v = 0; v < 3; ++v) atomicAdd(&confb[b * NN + tri[r * 3 + v]], s);
}

__global__ void conf_tanh_k(const float* __restrict__ confb, float* __restrict__ out_conf) {
    int i = blockIdx.x * blockDim.x + threadIdx.x;
    if (i < BNN) out_conf[i] = tanhf(confb[i]);
}

__global__ void final_kabsch_k(const float* __restrict__ Kg, float* __restrict__ outR) {
    int b = threadIdx.x;
    if (b >= BB) return;
    float Km[9], R[9];
#pragma unroll
    for (int i = 0; i < 9; ++i) Km[i] = Kg[b * 9 + i];
    kabsch3(Km, R);
#pragma unroll
    for (int i = 0; i < 9; ++i) outR[b * 9 + i] = R[i];
}

// =====================================================================
extern "C" void kernel_launch(void* const* d_in, const int* in_sizes, int n_in,
                              void* d_out, int out_size, void* d_ws, size_t ws_size,
                              hipStream_t stream) {
    // Inputs in setup_inputs() dict order, params flattened jax-pytree style
    // (sorted keys, depth-first):
    const float* descs    = (const float*)d_in[0];   // (4,2048,256)
    const float* kpts     = (const float*)d_in[1];   // (4,2048,2)
    const float* pts3d    = (const float*)d_in[2];   // (4,2048,3)
    const float* kpts_tp1 = (const float*)d_in[3];   // (4,2048,2)
    const float* intr     = (const float*)d_in[4];   // (4,4)
    const int*   tri      = (const int*)d_in[5];     // (4,4096,3)
    const float* g_att    = (const float*)d_in[6];   // gat/att   (4,64)
    const float* g_bias   = (const float*)d_in[7];   // gat/bias  (256)
    const float* g_we     = (const float*)d_in[8];   // gat/we    (3,256)
    const float* g_wl     = (const float*)d_in[9];   // gat/wl    (320,256)
    const float* g_wr     = (const float*)d_in[10];  // gat/wr    (320,256)
    const float* n_b      = (const float*)d_in[11];  // norm/b
    const float* n_g      = (const float*)d_in[12];  // norm/g
    const float* pe_l1b   = (const float*)d_in[13];  // pos_encoder/l1/b
    const float* pe_l1w   = (const float*)d_in[14];  // pos_encoder/l1/w (3,32)
    const float* pe_l2b   = (const float*)d_in[15];  // pos_encoder/l2/b
    const float* pe_l2w   = (const float*)d_in[16];  // pos_encoder/l2/w (32,64)
    const float* pe_ln1b  = (const float*)d_in[17];
    const float* pe_ln1g  = (const float*)d_in[18];
    const float* pe_ln2b  = (const float*)d_in[19];
    const float* pe_ln2g  = (const float*)d_in[20];
    const float* pj_b     = (const float*)d_in[21];  // projector/b
    const float* pj_w     = (const float*)d_in[22];  // projector/w (256,256)
    const float* rs_b     = (const float*)d_in[23];  // res_proj/b
    const float* rs_w     = (const float*)d_in[24];  // res_proj/w (320,256)
    const float* t1b      = (const float*)d_in[25];  // tri_head/l1/b
    const float* t1w      = (const float*)d_in[26];  // tri_head/l1/w (768,128)
    const float* t2b      = (const float*)d_in[27];  // tri_head/l2/b
    const float* t2w      = (const float*)d_in[28];  // tri_head/l2/w (128,128)
    const float* whb      = (const float*)d_in[29];  // tri_head/wh/b (1)
    const float* whw      = (const float*)d_in[30];  // tri_head/wh/w (128,1)
    (void)in_sizes; (void)n_in; (void)out_size; (void)ws_size;

    float* out = (float*)d_out;
    // output layout (flat, return order): R | cw | conf | edges | edge_attr
    float* out_R     = out;                         // 36
    float* out_cw    = out + 36;                    // 16384
    float* out_conf  = out + 36 + BTT;              // 8192
    float* out_edges = out + 36 + BTT + BNN;        // 2*EE (indices as f32)
    float* out_eattr = out_edges + 2 * EE;          // EE*3

    // workspace carve (~83 MB of f32)
    float* w = (float*)d_ws;
    float* xc  = w;                 w += (size_t)BNN * 320;
    float* xl  = w;                 w += (size_t)BNN * 256;
    float* xr  = w;                 w += (size_t)BNN * 256;
    float* rp  = w;                 w += (size_t)BNN * 256;
    float* agg = w;                 w += (size_t)BNN * 256;
    float* tb  = w;                 w += (size_t)BNN * 256;
    float* nf  = w;                 w += (size_t)BNN * 256;
    float* evals  = w;              w += (size_t)EE * 4;
    float* exvals = w;              w += (size_t)EE * 4;
    float* ssum   = w;              w += (size_t)BNN * 4;
    unsigned* mmax = (unsigned*)w;  w += (size_t)BNN * 4;
    int* srcdst = (int*)w;          w += (size_t)EE * 2;
    float* h1   = w;                w += (size_t)BTT * 128;
    float* h2   = w;                w += (size_t)BTT * 128;
    float* wtri = w;                w += BTT;
    float* Ktri = w;                w += (size_t)BTT * 9;
    float* xv   = w;                w += BTT;
    float* dens = w;                w += BTT;
    float* xvstar = w;              w += BB;
    float* Kg   = w;                w += BB * 9;
    float* confb = w;               w += BNN;

    const int TPB = 256;
#define GRID(n) dim3(((n) + TPB - 1) / TPB)

    // init accumulators (harness poisons ws once; we must re-init every call)
    fill_f32_k<<<GRID(BNN * 256), TPB, 0, stream>>>(agg, 0.f, BNN * 256);
    fill_f32_k<<<GRID(BNN * 4), TPB, 0, stream>>>(ssum, 0.f, BNN * 4);
    fill_u32_k<<<GRID(BNN * 4), TPB, 0, stream>>>(mmax, ORD_NEG_INF, BNN * 4);
    fill_f32_k<<<GRID(BNN), TPB, 0, stream>>>(confb, 0.f, BNN);
    fill_f32_k<<<GRID(BB * 9), TPB, 0, stream>>>(Kg, 0.f, BB * 9);

    // xc = [descs | pos_encoder(kpts, depth)]
    copy_desc_k<<<GRID(BNN * 256), TPB, 0, stream>>>(descs, xc);
    posenc_k<<<GRID(BNN), TPB, 0, stream>>>(kpts, pts3d, pe_l1w, pe_l1b, pe_ln1g, pe_ln1b,
                                            pe_l2w, pe_l2b, pe_ln2g, pe_ln2b, xc);

    // edges + edge_attr (also written to d_out)
    edge_build_k<<<GRID(BTT), TPB, 0, stream>>>(tri, kpts, out_edges, out_eattr, srcdst);

    // WMMA GEMMs: xl/xr/res_proj over xc (M=8192, K=320)
    gemm_wmma_dense_k<0, false><<<GRID((BNN / 16) * (256 / 16) * 32), TPB, 0, stream>>>(xc, g_wl, nullptr, xl, BNN, 256, 320);
    gemm_wmma_dense_k<0, false><<<GRID((BNN / 16) * (256 / 16) * 32), TPB, 0, stream>>>(xc, g_wr, nullptr, xr, BNN, 256, 320);
    gemm_wmma_dense_k<0, true><<<GRID((BNN / 16) * (256 / 16) * 32), TPB, 0, stream>>>(xc, rs_w, rs_b, rp, BNN, 256, 320);

    // GATv2 segment softmax + aggregation
    edge_e_k<<<GRID(EE * 32), TPB, 0, stream>>>(xl, xr, out_eattr, g_we, g_att, srcdst, evals, mmax);
    edge_softmax_k<<<GRID(EE * 4), TPB, 0, stream>>>(evals, mmax, srcdst, exvals, ssum);
    edge_agg_k<<<GRID(EE * 32), TPB, 0, stream>>>(xl, exvals, ssum, srcdst, agg);

    // node post-processing, projector GEMM -> node features
    node_post_k<<<GRID(BNN * 32), TPB, 0, stream>>>(agg, g_bias, n_g, n_b, rp, tb);
    gemm_wmma_dense_k<0, true><<<GRID((BNN / 16) * (256 / 16) * 32), TPB, 0, stream>>>(tb, pj_w, pj_b, nf, BNN, 256, 256);

    // triangle head: fused-gather GEMM (M=16384, K=768), then 128x128, then dot
    gemm_wmma_gather_k<<<GRID((BTT / 16) * (128 / 16) * 32), TPB, 0, stream>>>(nf, tri, t1w, t1b, h1);
    gemm_wmma_dense_k<1, true><<<GRID((BTT / 16) * (128 / 16) * 32), TPB, 0, stream>>>(h1, t2w, t2b, h2, BTT, 128, 128);
    dot_wh_k<<<GRID(BTT * 32), TPB, 0, stream>>>(h2, whw, whb, wtri);

    // per-triangle Kabsch + vanishing-x votes
    tri_K_k<<<GRID(BTT), TPB, 0, stream>>>(tri, pts3d, kpts_tp1, intr, Ktri, xv);

    // Gaussian voting (T x T per batch), weighted mean, consensus, global Kabsch
    dens_k<<<dim3(TT / 256, BB), TPB, 0, stream>>>(xv, wtri, dens);
    xvstar_k<<<dim3(BB), TPB, 0, stream>>>(xv, wtri, dens, xvstar);
    cw_k<<<GRID(BTT), TPB, 0, stream>>>(tri, xv, xvstar, wtri, Ktri, out_cw, Kg, confb);
    conf_tanh_k<<<GRID(BNN), TPB, 0, stream>>>(confb, out_conf);
    final_kabsch_k<<<1, 32, 0, stream>>>(Kg, out_R);
#undef GRID
}
